// KAL_Net_54279796687469
// MI455X (gfx1250) — matrix operand order repaired and checked by reading, not compiled, not run on hardware
//
#include <hip/hip_runtime.h>
#include <stdint.h>

typedef __attribute__((ext_vector_type(16))) __bf16 v16bf;
typedef __attribute__((ext_vector_type(8)))  float  v8f;

#define D_IN   50176
#define D_H    1024
#define D_H2   128
#define BATCH  256
#define PITCH  20   // LDS row pitch in dwords (16 data + 4 pad -> <=2-way conflicts)

__device__ __forceinline__ float silu_f(float x) {
    return x / (1.0f + __expf(-x));
}

// round-to-nearest-even fp32 -> 2x bf16 packed into a dword
__device__ __forceinline__ uint32_t pack2bf(float a, float b) {
    uint32_t ua = __float_as_uint(a), ub = __float_as_uint(b);
    ua = (ua + 0x7FFFu + ((ua >> 16) & 1u)) >> 16;
    ub = (ub + 0x7FFFu + ((ub >> 16) & 1u)) >> 16;
    return ua | (ub << 16);
}

// order-preserving float<->uint mapping for atomic min/max
__device__ __forceinline__ uint32_t encf(float f) {
    uint32_t u = __float_as_uint(f);
    return (u & 0x80000000u) ? ~u : (u | 0x80000000u);
}
__device__ __forceinline__ float decf(uint32_t u) {
    u = (u & 0x80000000u) ? (u & 0x7FFFFFFFu) : ~u;
    return __uint_as_float(u);
}

union FragU { uint32_t u[8]; uint4 q[2]; v16bf v; };

// 16-bit A 16x32 layout: lane half h, elems 0..7 -> K=h*8+e, elems 8..15 -> K=16+h*8+(e-8)
__device__ __forceinline__ v16bf load_a_frag(const uint32_t* row, int half) {
    FragU f;
    f.q[0] = *(const uint4*)(row + half * 4);
    f.q[1] = *(const uint4*)(row + 8 + half * 4);
    return f.v;
}
// 16-bit B 32x16 layout: lane half h, elems 0..7 -> K=h*8+e, elems 8..15 -> K=16+h*8+(e-8)
__device__ __forceinline__ v16bf load_b_frag(const uint32_t* row, int half) {
    FragU f;
    f.q[0] = *(const uint4*)(row + half * 4);
    f.q[1] = *(const uint4*)(row + 8 + half * 4);
    return f.v;
}

// ---------------------------------------------------------------- init + zero
__global__ void k_init(uint32_t* mm, float* h0, int nh) {
    int i = blockIdx.x * blockDim.x + threadIdx.x;
    if (i < nh) h0[i] = 0.0f;
    if (i < 4) mm[i] = (i & 1) ? 0u : 0xFFFFFFFFu;   // [min=+inf_enc, max=-inf_enc] x2
}

// ---------------------------------------------------------------- minmax of x
__global__ __launch_bounds__(256) void k_minmax_x(const float* __restrict__ x,
                                                  uint32_t* __restrict__ mm, int n4) {
    __shared__ float smn[256], smx[256];
    int tid = threadIdx.x;
    float mn = 3.402823466e38f, mx = -3.402823466e38f;
    for (int i = blockIdx.x * 256 + tid; i < n4; i += gridDim.x * 256) {
        float4 v = ((const float4*)x)[i];
        mn = fminf(mn, fminf(fminf(v.x, v.y), fminf(v.z, v.w)));
        mx = fmaxf(mx, fmaxf(fmaxf(v.x, v.y), fmaxf(v.z, v.w)));
    }
    smn[tid] = mn; smx[tid] = mx;
    __syncthreads();
    for (int off = 128; off > 0; off >>= 1) {
        if (tid < off) {
            smn[tid] = fminf(smn[tid], smn[tid + off]);
            smx[tid] = fmaxf(smx[tid], smx[tid + off]);
        }
        __syncthreads();
    }
    if (tid == 0) {
        atomicMin(&mm[0], encf(smn[0]));
        atomicMax(&mm[1], encf(smx[0]));
    }
}

// ------------------------------------------------- layer-0 fused split-K GEMM
// grid: (16 N-tiles of 64 cols) x (98 K-chunks of 512 features). 256 thr = 8 waves.
// Wave w: cols (w&3)*16, row-tiles [(w>>2)*8 .. +8) of the 256-row batch.
__global__ __launch_bounds__(256) void k_layer0_wmma(
    const float* __restrict__ x, const float* __restrict__ bw,
    const float* __restrict__ pw, const uint32_t* __restrict__ mm,
    float* __restrict__ h0)
{
    __shared__ __align__(16) uint32_t A_lds[BATCH * PITCH];  // 256 rows x 32 bf16
    __shared__ __align__(16) uint32_t B_lds[64 * PITCH];     // 64 cols  x 32 bf16

    const int tid  = threadIdx.x;
    const int lane = tid & 31;
    const int wave = tid >> 5;
    const int half = lane >> 4;
    const int l16  = lane & 15;
    const int n0   = blockIdx.x * 64;
    const int fs0  = blockIdx.y * 512;
    const int colSub  = wave & 3;
    const int rowBase = (wave >> 2) * 128;

    const float xmin = decf(mm[0]);
    const float xmax = decf(mm[1]);
    const float nsc  = 2.0f / (xmax - xmin);

    v8f acc[8];
#pragma unroll
    for (int rt = 0; rt < 8; ++rt)
#pragma unroll
        for (int e = 0; e < 8; ++e) acc[rt][e] = 0.0f;

    const float* xrow = x + (size_t)tid * D_IN;   // thread t stages batch row t
    uint32_t* arow = A_lds + tid * PITCH;
    const int jB = tid >> 2, partB = tid & 3;
    uint32_t* brow = B_lds + jB * PITCH + partB * 4;

    // ---------------- base branch: A = silu(x), B = base_w0 -----------------
    for (int ks = fs0; ks < fs0 + 512; ks += 32) {
        {
            const float4* px = (const float4*)(xrow + ks);
#pragma unroll
            for (int i = 0; i < 8; ++i) {
                float4 v = px[i];
                arow[2 * i]     = pack2bf(silu_f(v.x), silu_f(v.y));
                arow[2 * i + 1] = pack2bf(silu_f(v.z), silu_f(v.w));
            }
        }
        {
            const float* wr = bw + (size_t)(n0 + jB) * D_IN + ks + partB * 8;
            float4 w0 = ((const float4*)wr)[0];
            float4 w1 = ((const float4*)wr)[1];
            brow[0] = pack2bf(w0.x, w0.y);
            brow[1] = pack2bf(w0.z, w0.w);
            brow[2] = pack2bf(w1.x, w1.y);
            brow[3] = pack2bf(w1.z, w1.w);
        }
        __syncthreads();
        v16bf bf = load_b_frag(B_lds + (colSub * 16 + l16) * PITCH, half);
#pragma unroll
        for (int rt = 0; rt < 8; ++rt) {
            v16bf af = load_a_frag(A_lds + (rowBase + rt * 16 + l16) * PITCH, half);
            acc[rt] = __builtin_amdgcn_wmma_f32_16x16x32_bf16(
                false, af, false, bf, (short)0, acc[rt], false, false);
        }
        __syncthreads();
    }

    // -------- poly branch: A = Legendre(P0..P3)(xnorm), B = poly_w0 ---------
    for (int f8 = fs0; f8 < fs0 + 512; f8 += 8) {
        {
            const float4* px = (const float4*)(xrow + f8);
            float4 v0 = px[0], v1 = px[1];
            float xv[8] = {v0.x, v0.y, v0.z, v0.w, v1.x, v1.y, v1.z, v1.w};
#pragma unroll
            for (int i = 0; i < 8; ++i) {
                float xn  = (xv[i] - xmin) * nsc - 1.0f;
                float xn2 = xn * xn;
                float p2  = 1.5f * xn2 - 0.5f;
                float p3  = (2.5f * xn2 - 1.5f) * xn;
                arow[2 * i]     = pack2bf(1.0f, xn);
                arow[2 * i + 1] = pack2bf(p2, p3);
            }
        }
        {
            const float* wr = pw + (size_t)(n0 + jB) * (size_t)(D_IN * 4)
                              + 4 * f8 + partB * 8;
            float4 w0 = ((const float4*)wr)[0];
            float4 w1 = ((const float4*)wr)[1];
            brow[0] = pack2bf(w0.x, w0.y);
            brow[1] = pack2bf(w0.z, w0.w);
            brow[2] = pack2bf(w1.x, w1.y);
            brow[3] = pack2bf(w1.z, w1.w);
        }
        __syncthreads();
        v16bf bf = load_b_frag(B_lds + (colSub * 16 + l16) * PITCH, half);
#pragma unroll
        for (int rt = 0; rt < 8; ++rt) {
            v16bf af = load_a_frag(A_lds + (rowBase + rt * 16 + l16) * PITCH, half);
            acc[rt] = __builtin_amdgcn_wmma_f32_16x16x32_bf16(
                false, af, false, bf, (short)0, acc[rt], false, false);
        }
        __syncthreads();
    }

    // split-K combine (fp32, into L2-resident 1MB accumulator)
    const int n = n0 + colSub * 16 + l16;
#pragma unroll
    for (int rt = 0; rt < 8; ++rt)
#pragma unroll
        for (int e = 0; e < 8; ++e) {
            int m = rowBase + rt * 16 + e + 8 * half;   // C/D layout: VGPR e -> M=e+8*half
            atomicAdd(h0 + (size_t)m * D_H + n, acc[rt][e]);
        }
}

// --------------------------- layer-0 epilogue: LayerNorm + silu + minmax(h1)
__global__ __launch_bounds__(256) void k_layer0_post(
    const float* __restrict__ h0, const float* __restrict__ g,
    const float* __restrict__ b, float* __restrict__ h1,
    uint32_t* __restrict__ mm)
{
    __shared__ float r0[256], r1[256];
    int row = blockIdx.x, tid = threadIdx.x;
    float4 v = ((const float4*)(h0 + (size_t)row * D_H))[tid];
    r0[tid] = v.x + v.y + v.z + v.w;
    r1[tid] = v.x * v.x + v.y * v.y + v.z * v.z + v.w * v.w;
    __syncthreads();
    for (int off = 128; off > 0; off >>= 1) {
        if (tid < off) { r0[tid] += r0[tid + off]; r1[tid] += r1[tid + off]; }
        __syncthreads();
    }
    float mu  = r0[0] * (1.0f / 1024.0f);
    float var = r1[0] * (1.0f / 1024.0f) - mu * mu;
    float inv = rsqrtf(var + 1e-5f);
    float4 gg = ((const float4*)g)[tid];
    float4 bb = ((const float4*)b)[tid];
    float o0 = silu_f((v.x - mu) * inv * gg.x + bb.x);
    float o1 = silu_f((v.y - mu) * inv * gg.y + bb.y);
    float o2 = silu_f((v.z - mu) * inv * gg.z + bb.z);
    float o3 = silu_f((v.w - mu) * inv * gg.w + bb.w);
    ((float4*)(h1 + (size_t)row * D_H))[tid] = make_float4(o0, o1, o2, o3);
    float mn = fminf(fminf(o0, o1), fminf(o2, o3));
    float mx = fmaxf(fmaxf(o0, o1), fmaxf(o2, o3));
    __syncthreads();
    r0[tid] = mn; r1[tid] = mx;
    __syncthreads();
    for (int off = 128; off > 0; off >>= 1) {
        if (tid < off) {
            r0[tid] = fminf(r0[tid], r0[tid + off]);
            r1[tid] = fmaxf(r1[tid], r1[tid + off]);
        }
        __syncthreads();
    }
    if (tid == 0) {
        atomicMin(&mm[2], encf(r0[0]));
        atomicMax(&mm[3], encf(r1[0]));
    }
}

// ---------------- layer-1 (WMMA) + LayerNorm + silu + logits + softmax ------
// grid: 16 blocks (one per 16-row tile). 8 waves own 16 output cols each (128 total).
__global__ __launch_bounds__(256) void k_layer1_out(
    const float* __restrict__ h1, const float* __restrict__ bw,
    const float* __restrict__ pw, const float* __restrict__ g,
    const float* __restrict__ b, const float* __restrict__ ow,
    const float* __restrict__ ob, const uint32_t* __restrict__ mm,
    float* __restrict__ out)
{
    __shared__ __align__(16) uint32_t A_lds[16 * PITCH];
    __shared__ __align__(16) uint32_t B_lds[D_H2 * PITCH];
    __shared__ float htile[16][132];
    __shared__ float rowMu[16], rowInv[16];

    const int tid  = threadIdx.x;
    const int lane = tid & 31;
    const int wave = tid >> 5;
    const int half = lane >> 4;
    const int l16  = lane & 15;
    const int r0b  = blockIdx.x * 16;

    const float hmin = decf(mm[2]);
    const float hmax = decf(mm[3]);
    const float nsc  = 2.0f / (hmax - hmin);

    v8f acc;
#pragma unroll
    for (int e = 0; e < 8; ++e) acc[e] = 0.0f;

    const int jB = tid >> 1, partB = tid & 1;
    uint32_t* brow = B_lds + jB * PITCH + partB * 8;

    // base branch over K=1024
    for (int ks = 0; ks < D_H; ks += 32) {
        {
            int r = tid >> 4, c = (tid & 15) * 2;
            const float* p = h1 + (size_t)(r0b + r) * D_H + ks + c;
            A_lds[r * PITCH + (c >> 1)] = pack2bf(silu_f(p[0]), silu_f(p[1]));
        }
        {
            const float* wr = bw + (size_t)jB * D_H + ks + partB * 16;
#pragma unroll
            for (int i = 0; i < 4; ++i) {
                float4 w = ((const float4*)wr)[i];
                brow[2 * i]     = pack2bf(w.x, w.y);
                brow[2 * i + 1] = pack2bf(w.z, w.w);
            }
        }
        __syncthreads();
        v16bf bf = load_b_frag(B_lds + (wave * 16 + l16) * PITCH, half);
        v16bf af = load_a_frag(A_lds + l16 * PITCH, half);
        acc = __builtin_amdgcn_wmma_f32_16x16x32_bf16(
            false, af, false, bf, (short)0, acc, false, false);
        __syncthreads();
    }
    // poly branch over K'=4096 (8 features per step)
    for (int f8 = 0; f8 < D_H; f8 += 8) {
        if (tid < 128) {
            int r = tid >> 3, fl = tid & 7;
            float xv  = h1[(size_t)(r0b + r) * D_H + f8 + fl];
            float xn  = (xv - hmin) * nsc - 1.0f;
            float xn2 = xn * xn;
            float p2  = 1.5f * xn2 - 0.5f;
            float p3  = (2.5f * xn2 - 1.5f) * xn;
            uint32_t* arow = A_lds + r * PITCH + fl * 2;
            arow[0] = pack2bf(1.0f, xn);
            arow[1] = pack2bf(p2, p3);
        }
        {
            const float* wr = pw + (size_t)jB * (D_H * 4) + 4 * f8 + partB * 16;
#pragma unroll
            for (int i = 0; i < 4; ++i) {
                float4 w = ((const float4*)wr)[i];
                brow[2 * i]     = pack2bf(w.x, w.y);
                brow[2 * i + 1] = pack2bf(w.z, w.w);
            }
        }
        __syncthreads();
        v16bf bf = load_b_frag(B_lds + (wave * 16 + l16) * PITCH, half);
        v16bf af = load_a_frag(A_lds + l16 * PITCH, half);
        acc = __builtin_amdgcn_wmma_f32_16x16x32_bf16(
            false, af, false, bf, (short)0, acc, false, false);
        __syncthreads();
    }

    {   // spill accumulator tile to LDS in (row, col) order
        int n = wave * 16 + l16;
#pragma unroll
        for (int e = 0; e < 8; ++e) htile[e + 8 * half][n] = acc[e];
    }
    __syncthreads();
    if (tid < 16) {   // per-row LayerNorm stats over 128 (tiny)
        float s = 0.0f, s2 = 0.0f;
        for (int k = 0; k < D_H2; ++k) { float h = htile[tid][k]; s += h; s2 += h * h; }
        float mu = s * (1.0f / 128.0f);
        float var = s2 * (1.0f / 128.0f) - mu * mu;
        rowMu[tid]  = mu;
        rowInv[tid] = rsqrtf(var + 1e-5f);
    }
    __syncthreads();
    for (int i = tid; i < 16 * D_H2; i += 256) {
        int r = i >> 7, k = i & 127;
        float h = (htile[r][k] - rowMu[r]) * rowInv[r] * g[k] + b[k];
        htile[r][k] = silu_f(h);
    }
    __syncthreads();
    if (tid < 16) {   // logits (2 dots of 128) + softmax
        float l0 = ob[0], l1 = ob[1];
        for (int k = 0; k < D_H2; ++k) {
            float h = htile[tid][k];
            l0 += h * ow[k];
            l1 += h * ow[D_H2 + k];
        }
        float m  = fmaxf(l0, l1);
        float e0 = __expf(l0 - m), e1 = __expf(l1 - m);
        float inv = 1.0f / (e0 + e1);
        out[(r0b + tid) * 2 + 0] = e0 * inv;
        out[(r0b + tid) * 2 + 1] = e1 * inv;
    }
}

// ---------------------------------------------------------------------------
extern "C" void kernel_launch(void* const* d_in, const int* in_sizes, int n_in,
                              void* d_out, int out_size, void* d_ws, size_t ws_size,
                              hipStream_t stream) {
    const float* x   = (const float*)d_in[0];
    const float* bw0 = (const float*)d_in[1];
    const float* pw0 = (const float*)d_in[2];
    const float* g0  = (const float*)d_in[3];
    const float* b0  = (const float*)d_in[4];
    const float* bw1 = (const float*)d_in[5];
    const float* pw1 = (const float*)d_in[6];
    const float* g1  = (const float*)d_in[7];
    const float* b1  = (const float*)d_in[8];
    const float* ow  = (const float*)d_in[9];
    const float* ob  = (const float*)d_in[10];
    float* out = (float*)d_out;

    uint32_t* mm = (uint32_t*)d_ws;                       // 4 encoded min/max slots
    float* h0 = (float*)((char*)d_ws + 256);              // [256,1024] accumulator
    float* h1 = h0 + BATCH * D_H;                         // [256,1024] layer-1 input

    const int nh = BATCH * D_H;
    k_init<<<(nh + 255) / 256, 256, 0, stream>>>(mm, h0, nh);
    k_minmax_x<<<512, 256, 0, stream>>>(x, mm, (BATCH * D_IN) / 4);
    k_layer0_wmma<<<dim3(16, 98), 256, 0, stream>>>(x, bw0, pw0, mm, h0);
    k_layer0_post<<<BATCH, 256, 0, stream>>>(h0, g0, b0, h1, mm);
    k_layer1_out<<<16, 256, 0, stream>>>(h1, bw1, pw1, g1, b1, ow, ob, mm, out);
}
